// KNRM_28149215658319
// MI455X (gfx1250) — matrix operand order repaired and checked by reading, not compiled, not run on hardware
//
#include <hip/hip_runtime.h>
#include <hip/hip_bf16.h>
#include <math.h>

// ---------------- types for CDNA5 WMMA ----------------
typedef __bf16  v16bf __attribute__((ext_vector_type(16)));
typedef float   v8f   __attribute__((ext_vector_type(8)));

#define VOCAB 100000
#define DIMD  256
#define BSZ   512
#define QL    32
#define DL    256
#define NKER  21
#define MMSTR 260          // padded LDS row stride (dwords); 260 % 64 == 4

// workspace layout
#define TABLE_BYTES ((size_t)VOCAB * DIMD * 2)            // bf16 normalized table
#define KM_OFF      TABLE_BYTES                            // float km[2][BSZ][NKER]

#define CF2 (-72.13475204444817f)   // (-0.5 / 0.1^2) / ln(2)

// -------------------------------------------------------------------------
// Kernel 1: L2-normalize every vocab row (f32 math) and store bf16 copy.
// One 32-lane wave per row; 8 rows per 256-thread block.
// -------------------------------------------------------------------------
__global__ void normalize_table(const float* __restrict__ emb,
                                __bf16* __restrict__ table) {
    const int lane = threadIdx.x & 31;
    const int rowInBlk = threadIdx.x >> 5;
    const int row = blockIdx.x * 8 + rowInBlk;
    if (row >= VOCAB) return;

    const float* src = emb + (size_t)row * DIMD + lane * 8;
    float4 a = *reinterpret_cast<const float4*>(src);
    float4 b = *reinterpret_cast<const float4*>(src + 4);

    float ss = a.x*a.x + a.y*a.y + a.z*a.z + a.w*a.w
             + b.x*b.x + b.y*b.y + b.z*b.z + b.w*b.w;
    // wave32 butterfly reduction
    #pragma unroll
    for (int m = 16; m >= 1; m >>= 1) ss += __shfl_xor(ss, m, 32);

    const float inv = 1.0f / fmaxf(sqrtf(ss), 1e-8f);

    union { __bf16 h[8]; uint4 u; } pk;
    pk.h[0] = (__bf16)(a.x * inv); pk.h[1] = (__bf16)(a.y * inv);
    pk.h[2] = (__bf16)(a.z * inv); pk.h[3] = (__bf16)(a.w * inv);
    pk.h[4] = (__bf16)(b.x * inv); pk.h[5] = (__bf16)(b.y * inv);
    pk.h[6] = (__bf16)(b.z * inv); pk.h[7] = (__bf16)(b.w * inv);
    *reinterpret_cast<uint4*>(table + (size_t)row * DIMD + lane * 8) = pk.u;
}

// ---- fragment loaders, per CDNA5 ISA 16-bit VGPR layouts (wave32) --------
// A (16x32, MxK): lanes 0-15 row m hold K {0..7,16..23}; lanes 16-31 hold
// K {8..15,24..31}  -> two contiguous 16B chunks per lane.
__device__ __forceinline__ v16bf load_fragA(const __bf16* __restrict__ rowp,
                                            int k0, int hi) {
    union { uint4 u[2]; v16bf v; } f;
    f.u[0] = *reinterpret_cast<const uint4*>(rowp + k0 + hi * 8);
    f.u[1] = *reinterpret_cast<const uint4*>(rowp + k0 + 16 + hi * 8);
    return f.v;
}
// B (32x16, KxN): lanes 0-15 col n hold K {0..7,16..23}; lanes 16-31 hold
// K {8..15,24..31} -> two 16B chunks per lane.  B[k][n] = dn[n][k] (row-major dn).
__device__ __forceinline__ v16bf load_fragB(const __bf16* __restrict__ rowp,
                                            int k0, int hi) {
    union { uint4 u[2]; v16bf v; } f;
    const __bf16* p = rowp + k0 + hi * 8;
    f.u[0] = *reinterpret_cast<const uint4*>(p);
    f.u[1] = *reinterpret_cast<const uint4*>(p + 16);
    return f.v;
}

// -------------------------------------------------------------------------
// Kernel 2: per (batch, pair) workgroup.  8 waves: WMMA cosine matrix
// 32x256 (K=256) -> LDS, then fused Gaussian kernel pooling + log1p.
// -------------------------------------------------------------------------
__global__ void __launch_bounds__(256)
knrm_batch(const int* __restrict__ q1, const int* __restrict__ d1,
           const int* __restrict__ q2, const int* __restrict__ d2,
           const __bf16* __restrict__ table,
           float* __restrict__ km_out) {
    __shared__ float mm_s[QL * MMSTR];       // ~32.5 KB, padded stride
    __shared__ float bins[QL * NKER];        // 2.6 KB
    __shared__ int   qids[QL];
    __shared__ int   dids[DL];

    const int b    = blockIdx.x;
    const int pair = blockIdx.y;
    const int tid  = threadIdx.x;
    const int* q = (pair == 0) ? q1 : q2;
    const int* d = (pair == 0) ? d1 : d2;

    dids[tid] = d[b * DL + tid];
    if (tid < QL)  qids[tid] = q[b * QL + tid];
    for (int i = tid; i < QL * NKER; i += 256) bins[i] = 0.0f;
    __syncthreads();

    // ---------------- WMMA GEMM phase ----------------
    const int lane   = tid & 31;
    const int wid    = tid >> 5;         // 0..7
    const int rl     = lane & 15;
    const int hi     = lane >> 4;
    const int m_tile = wid & 1;          // 2 row tiles of 16
    const int group  = wid >> 1;         // 4 waves across N

    const __bf16* arow = table + (size_t)qids[m_tile * 16 + rl] * DIMD;
    v16bf afr[8];
    #pragma unroll
    for (int ks = 0; ks < 8; ++ks) afr[ks] = load_fragA(arow, ks * 32, hi);

    #pragma unroll
    for (int j = 0; j < 4; ++j) {
        const int nt = group * 4 + j;                    // 16 col tiles
        const __bf16* brow = table + (size_t)dids[nt * 16 + rl] * DIMD;
        v8f acc = {};
        #pragma unroll
        for (int ks = 0; ks < 8; ++ks) {
            v16bf bfr = load_fragB(brow, ks * 32, hi);
            acc = __builtin_amdgcn_wmma_f32_16x16x32_bf16(
                      false, afr[ks], false, bfr, (short)0, acc, false, false);
        }
        // C layout: lane l, VGPR v -> M = v + 8*(l>>4), N = l&15
        #pragma unroll
        for (int v = 0; v < 8; ++v) {
            const int m_loc = m_tile * 16 + v + 8 * hi;
            const int n_loc = nt * 16 + rl;
            mm_s[m_loc * MMSTR + n_loc] = acc[v];
        }
    }
    __syncthreads();

    // ---------------- Gaussian soft-binning pooling ----------------
    // lane = row (conflict-free across a wave), chunk = tid>>5 covers 32 cols.
    // exp(-0.5(mm-mu)^2/s^2) = exp2(a + P_k + b*mu_k) with a = CF2*mm^2,
    // b = -2*CF2*mm, P_k = CF2*mu_k^2  -> 2 VALU + 1 v_exp per kernel.
    // Exact kernel (sigma=1e-3) == token-id match (else underflows to 0).
    float mu[20], P[20];
    #pragma unroll
    for (int k = 0; k < 20; ++k) {
        mu[k] = (2 * k + 1) * 0.05f - 1.0f;
        P[k]  = CF2 * mu[k] * mu[k];
    }
    const int m   = tid & 31;
    const int c   = tid >> 5;
    const int qid = qids[m];
    float acc[NKER];
    #pragma unroll
    for (int k = 0; k < NKER; ++k) acc[k] = 0.0f;

    const float4* mrow = reinterpret_cast<const float4*>(&mm_s[m * MMSTR + c * 32]);
    for (int v4 = 0; v4 < 8; ++v4) {          // keep rolled: compact hot loop
        float4 mv = mrow[v4];
        float vals[4] = {mv.x, mv.y, mv.z, mv.w};
        #pragma unroll
        for (int i = 0; i < 4; ++i) {
            const int n = c * 32 + v4 * 4 + i;
            const bool match = (dids[n] == qid);
            const float mm = match ? 1.0f : vals[i];
            const float a  = CF2 * mm * mm;
            const float bb = -2.0f * CF2 * mm;
            #pragma unroll
            for (int k = 0; k < 20; ++k)
                acc[k] += exp2f(fmaf(bb, mu[k], a + P[k]));
            acc[20] += match ? 1.0f : 0.0f;
        }
    }
    #pragma unroll
    for (int k = 0; k < NKER; ++k)
        atomicAdd(&bins[m * NKER + k], acc[k]);   // ds_add_f32
    __syncthreads();

    if (tid < NKER) {
        float s = 0.0f;
        #pragma unroll
        for (int mm2 = 0; mm2 < QL; ++mm2) s += log1pf(bins[mm2 * NKER + tid]);
        km_out[((size_t)pair * BSZ + b) * NKER + tid] = s;
    }
}

// -------------------------------------------------------------------------
// Kernel 3: tiny MLP (21->10->5->1) for both pairs + sigmoid(l1-l2).
// -------------------------------------------------------------------------
__device__ __forceinline__ float mlp_eval(const float* km,
                                          const float* W0, const float* b0,
                                          const float* W1, const float* b1,
                                          const float* W2, const float* b2) {
    float h0[10];
    #pragma unroll
    for (int i = 0; i < 10; ++i) {
        float s = b0[i];
        #pragma unroll
        for (int j = 0; j < NKER; ++j) s = fmaf(km[j], W0[i * NKER + j], s);
        h0[i] = fmaxf(s, 0.0f);
    }
    float h1[5];
    #pragma unroll
    for (int i = 0; i < 5; ++i) {
        float s = b1[i];
        #pragma unroll
        for (int j = 0; j < 10; ++j) s = fmaf(h0[j], W1[i * 10 + j], s);
        h1[i] = fmaxf(s, 0.0f);
    }
    float s = b2[0];
    #pragma unroll
    for (int j = 0; j < 5; ++j) s = fmaf(h1[j], W2[j], s);
    return s;
}

__global__ void mlp_final(const float* __restrict__ km,
                          const float* __restrict__ W0, const float* __restrict__ b0,
                          const float* __restrict__ W1, const float* __restrict__ b1,
                          const float* __restrict__ W2, const float* __restrict__ b2,
                          float* __restrict__ out) {
    const int b = blockIdx.x * blockDim.x + threadIdx.x;
    if (b >= BSZ) return;
    float km1[NKER], km2[NKER];
    #pragma unroll
    for (int k = 0; k < NKER; ++k) {
        km1[k] = km[(size_t)b * NKER + k];
        km2[k] = km[((size_t)BSZ + b) * NKER + k];
    }
    const float l1 = mlp_eval(km1, W0, b0, W1, b1, W2, b2);
    const float l2 = mlp_eval(km2, W0, b0, W1, b1, W2, b2);
    out[b] = 1.0f / (1.0f + __expf(-(l1 - l2)));
}

// -------------------------------------------------------------------------
extern "C" void kernel_launch(void* const* d_in, const int* in_sizes, int n_in,
                              void* d_out, int out_size, void* d_ws, size_t ws_size,
                              hipStream_t stream) {
    const int*   q1  = (const int*)  d_in[0];
    const int*   d1  = (const int*)  d_in[1];
    const int*   q2  = (const int*)  d_in[2];
    const int*   d2  = (const int*)  d_in[3];
    const float* emb = (const float*)d_in[4];
    const float* W0  = (const float*)d_in[5];
    const float* b0  = (const float*)d_in[6];
    const float* W1  = (const float*)d_in[7];
    const float* b1  = (const float*)d_in[8];
    const float* W2  = (const float*)d_in[9];
    const float* b2  = (const float*)d_in[10];
    float* out = (float*)d_out;

    __bf16* table = (__bf16*)d_ws;
    float*  km    = (float*)((char*)d_ws + KM_OFF);

    normalize_table<<<VOCAB / 8, 256, 0, stream>>>(emb, table);
    knrm_batch<<<dim3(BSZ, 2), 256, 0, stream>>>(q1, d1, q2, d2, table, km);
    mlp_final<<<2, 256, 0, stream>>>(km, W0, b0, W1, b1, W2, b2, out);
}
